// EventTransformerClassifier_27118423507236
// MI455X (gfx1250) — matrix operand order repaired and checked by reading, not compiled, not run on hardware
//
#include <hip/hip_runtime.h>

// ---------------------------------------------------------------------------
// EventTransformerClassifier for MI455X (gfx1250), wave32 + WMMA bf16.
// B=8 L=512 D=512 H=8 DH=64 NL=6 F=2048 M=1024
// GEMM A-tiles staged by the Tensor Data Mover (TENSORcnt); attention K-tiles
// staged by async global->LDS (ASYNCcnt); softmax reductions in DPP VALU.
// ---------------------------------------------------------------------------

#define BB 8
#define LL 512
#define DD 512
#define HH 8
#define DH 64
#define NLAYER 6
#define FF 2048
#define NTOK (BB * LL)      // 4096
#define MREL 1024

typedef __bf16 bf16;
typedef float v8f  __attribute__((ext_vector_type(8)));
typedef bf16  v16bf __attribute__((ext_vector_type(16)));
typedef unsigned int u32x4 __attribute__((ext_vector_type(4)));
typedef int i32x4 __attribute__((ext_vector_type(4)));
typedef int i32x8 __attribute__((ext_vector_type(8)));

union FragBF { v16bf v; u32x4 u[2]; };
union Pack8  { u32x4 u; bf16 e[8]; };

static __device__ __forceinline__ v8f wmma_bf16(v16bf a, v16bf b, v8f c) {
    // D = A(16x32) * B(32x16) + C, f32 accumulate
    return __builtin_amdgcn_wmma_f32_16x16x32_bf16(
        /*neg_a=*/false, a, /*neg_b=*/false, b,
        /*c_mod=*/(short)0, c, /*reuse_a=*/false, /*reuse_b=*/false);
}

// A fragment: 16(M) x 32(K), row-major source with given element stride.
// lane<16: row=lane, K in {0..7,16..23}; lane>=16: row=lane-16, K in {8..15,24..31}
static __device__ __forceinline__ v16bf load_frag_a(const bf16* base, int stride) {
    const int lane = threadIdx.x & 31;
    const int hl   = lane >> 4;
    const int row  = lane & 15;
    const bf16* p = base + row * stride + hl * 8;
    FragBF f;
    f.u[0] = *(const u32x4*)(p);
    f.u[1] = *(const u32x4*)(p + 16);
    return f.v;
}

// B fragment: 32(K) x 16(N). Source is N-major [N][K] (transposed) with given
// stride. lane<16: col=lane, K in {0..7,16..23}; lane>=16: col=lane-16, K in {8..15,24..31}
static __device__ __forceinline__ v16bf load_frag_b(const bf16* base, int stride) {
    const int lane = threadIdx.x & 31;
    const int hl   = lane >> 4;
    const int n    = lane & 15;
    const bf16* p = base + n * stride + hl * 8;
    FragBF f;
    f.u[0] = *(const u32x4*)(p);
    f.u[1] = *(const u32x4*)(p + 16);
    return f.v;
}

// Async global->LDS 16B copy per lane (CDNA5 ASYNCcnt path).
static __device__ __forceinline__ void async_copy_b128(unsigned ldsOff, const void* gaddr) {
    asm volatile("global_load_async_to_lds_b128 %0, %1, off"
                 :: "v"(ldsOff), "v"(gaddr) : "memory");
}
static __device__ __forceinline__ void async_wait0() {
    asm volatile("s_wait_asynccnt 0" ::: "memory");
}
static __device__ __forceinline__ unsigned lds_off(const void* p) {
    return (unsigned)(size_t)p;   // shared aperture: addr[31:0] == LDS offset
}

// Tensor Data Mover: load a 2D tile [rows x width] of bf16 (width elems,
// element stride 'strideElems' between rows) from global into LDS.
// Must be executed by exactly one wave (TDM ignores EXEC).
static __device__ __forceinline__ void tdm_load_tile_bf16(
        unsigned ldsByteOff, const void* gaddr, int width, int rows, int strideElems) {
    unsigned long long ga = (unsigned long long)gaddr;
    u32x4 g0;
    g0[0] = 1u;                                             // count=1 (valid D#)
    g0[1] = ldsByteOff;                                     // lds_addr
    g0[2] = (unsigned)(ga & 0xFFFFFFFFull);                 // global_addr[31:0]
    g0[3] = (unsigned)((ga >> 32) & 0x01FFFFFFull) | (2u << 30);  // [56:32] + type=2
    i32x8 g1;
    g1[0] = 1 << 16;                                        // data_size = 2 bytes
    g1[1] = (width & 0xFFFF) << 16;                         // tensor_dim0[15:0]
    g1[2] = (rows & 0xFFFF) << 16;                          // tensor_dim1[15:0]
    g1[3] = (width & 0xFFFF) << 16;                         // tile_dim0
    g1[4] = rows & 0xFFFF;                                  // tile_dim1
    g1[5] = strideElems;                                    // tensor_dim0_stride[31:0]
    g1[6] = 0;
    g1[7] = 0;
    i32x4 z4 = {};
#if __clang_major__ >= 23
    i32x8 z8 = {};
    __builtin_amdgcn_tensor_load_to_lds(g0, g1, z4, z4, z8, 0);
#else
    __builtin_amdgcn_tensor_load_to_lds(g0, g1, z4, z4, 0);
#endif
}

// 16-lane butterfly reductions in pure VALU via DPP (stays within each
// 16-lane half of the wave; matches the WMMA C-fragment row grouping).
template <int CTRL>
static __device__ __forceinline__ float dpp_bfly(float x) {
    return __int_as_float(
        __builtin_amdgcn_mov_dpp(__float_as_int(x), CTRL, 0xF, 0xF, true));
}
static __device__ __forceinline__ float red16_max(float x) {
    x = fmaxf(x, dpp_bfly<0xB1>(x));   // quad_perm [1,0,3,2]  (xor 1)
    x = fmaxf(x, dpp_bfly<0x4E>(x));   // quad_perm [2,3,0,1]  (xor 2)
    x = fmaxf(x, dpp_bfly<0x141>(x));  // row_half_mirror      (xor 4 eff.)
    x = fmaxf(x, dpp_bfly<0x140>(x));  // row_mirror           (xor 8 eff.)
    return x;
}
static __device__ __forceinline__ float red16_sum(float x) {
    x += dpp_bfly<0xB1>(x);
    x += dpp_bfly<0x4E>(x);
    x += dpp_bfly<0x141>(x);
    x += dpp_bfly<0x140>(x);
    return x;
}

// ---------------------------------------------------------------------------
// f32 -> bf16 conversion (weights)
// ---------------------------------------------------------------------------
__global__ void f32_to_bf16_kernel(const float* __restrict__ src,
                                   bf16* __restrict__ dst, int n) {
    int i = blockIdx.x * blockDim.x + threadIdx.x;
    if (i < n) dst[i] = (bf16)src[i];
}

// ---------------------------------------------------------------------------
// relative-position bias table [L, L]
// ---------------------------------------------------------------------------
__global__ void bias_tab_kernel(const float* __restrict__ rel_table,
                                float* __restrict__ tab) {
    int q = blockIdx.x;
    for (int k = threadIdx.x; k < LL; k += blockDim.x) {
        int r = q - k;
        r = r < -(MREL - 1) ? -(MREL - 1) : (r > (MREL - 1) ? (MREL - 1) : r);
        tab[q * LL + k] = rel_table[r + (MREL - 1)];
    }
}

// ---------------------------------------------------------------------------
// Embedding front-end: xin[13] -> @W1+b1 -> LN -> ReLU -> bf16 hn
// one block (256 thr) per token
// ---------------------------------------------------------------------------
__global__ __launch_bounds__(256)
void embed_kernel(const int* __restrict__ ev_ids, const float* __restrict__ tdel,
                  const float* __restrict__ cont, const int* __restrict__ pkg,
                  const float* __restrict__ emb, const float* __restrict__ W1,
                  const float* __restrict__ b1, const float* __restrict__ g1,
                  const float* __restrict__ be1, bf16* __restrict__ out) {
    const int t = blockIdx.x;   // token
    __shared__ float xin[16];
    __shared__ float h[DD];
    __shared__ float red[256];
    const int tid = threadIdx.x;
    if (tid < 13) {
        float v;
        if (tid < 8) {
            int eid = ev_ids[t];
            v = emb[eid * 8 + tid];
        } else if (tid < 11) {
            v = cont[t * 3 + (tid - 8)];
        } else if (tid == 11) {
            v = (float)pkg[t];
        } else {
            v = tdel[t];
        }
        xin[tid] = v;
    }
    __syncthreads();
    for (int j = tid; j < DD; j += 256) {
        float acc = b1[j];
#pragma unroll
        for (int i = 0; i < 13; ++i) acc += xin[i] * W1[i * DD + j];
        h[j] = acc;
    }
    __syncthreads();
    float v0 = h[tid], v1 = h[tid + 256];
    red[tid] = v0 + v1;
    __syncthreads();
    for (int o = 128; o > 0; o >>= 1) {
        if (tid < o) red[tid] += red[tid + o];
        __syncthreads();
    }
    float mu = red[0] * (1.0f / DD);
    __syncthreads();
    float d0 = v0 - mu, d1 = v1 - mu;
    red[tid] = d0 * d0 + d1 * d1;
    __syncthreads();
    for (int o = 128; o > 0; o >>= 1) {
        if (tid < o) red[tid] += red[tid + o];
        __syncthreads();
    }
    float rstd = rsqrtf(red[0] * (1.0f / DD) + 1e-5f);
    float y0 = d0 * rstd * g1[tid] + be1[tid];
    float y1 = d1 * rstd * g1[tid + 256] + be1[tid + 256];
    out[(size_t)t * DD + tid]       = (bf16)(y0 > 0.f ? y0 : 0.f);
    out[(size_t)t * DD + tid + 256] = (bf16)(y1 > 0.f ? y1 : 0.f);
}

// ---------------------------------------------------------------------------
// LayerNorm: f32 x[token,512] -> bf16 out (affine)
// ---------------------------------------------------------------------------
__global__ __launch_bounds__(256)
void ln_to_bf16_kernel(const float* __restrict__ x, const float* __restrict__ g,
                       const float* __restrict__ b, bf16* __restrict__ out) {
    const int t = blockIdx.x;
    const int tid = threadIdx.x;
    __shared__ float red[256];
    const float* xr = x + (size_t)t * DD;
    float v0 = xr[tid], v1 = xr[tid + 256];
    red[tid] = v0 + v1;
    __syncthreads();
    for (int o = 128; o > 0; o >>= 1) {
        if (tid < o) red[tid] += red[tid + o];
        __syncthreads();
    }
    float mu = red[0] * (1.0f / DD);
    __syncthreads();
    float d0 = v0 - mu, d1 = v1 - mu;
    red[tid] = d0 * d0 + d1 * d1;
    __syncthreads();
    for (int o = 128; o > 0; o >>= 1) {
        if (tid < o) red[tid] += red[tid + o];
        __syncthreads();
    }
    float rstd = rsqrtf(red[0] * (1.0f / DD) + 1e-5f);
    out[(size_t)t * DD + tid]       = (bf16)(d0 * rstd * g[tid] + b[tid]);
    out[(size_t)t * DD + tid + 256] = (bf16)(d1 * rstd * g[tid + 256] + b[tid + 256]);
}

// ---------------------------------------------------------------------------
// WMMA GEMM: C[M,N] = A[M,K](bf16) * W[K,N](bf16) + bias
// block tile 128x64, 8 waves (4x2), each wave 32x32; K-step 64 (8 WMMA/step).
// A tile (128x64) staged by ONE tensor_load_to_lds (TDM) issued by wave 0;
// W tile transposed via registers. MODE 0: outF = val; 1: outF += val;
// 2: outB = bf16(val); 3: outB = bf16(relu(val))
// ---------------------------------------------------------------------------
template <int MODE>
__global__ __launch_bounds__(256)
void gemm_bf16_kernel(const bf16* __restrict__ A, const bf16* __restrict__ W,
                      const float* __restrict__ bias, float* __restrict__ outF,
                      bf16* __restrict__ outB, int M, int N, int K) {
    __shared__ __align__(16) bf16 sA[128 * 64];   // [row][k], stride 64
    __shared__ __align__(16) bf16 sBt[64 * 64];   // [n][k],   stride 64
    const int tid = threadIdx.x;
    const int wave = tid >> 5, lane = tid & 31;
    const int waveM = wave & 3, waveN = wave >> 2;
    const int mBase = blockIdx.x * 128;
    const int nBase = blockIdx.y * 64;

    v8f c[2][2] = {};

    for (int k0 = 0; k0 < K; k0 += 64) {
        // one TDM descriptor stages the whole 128x64 A tile into LDS
        if (wave == 0) {
            tdm_load_tile_bf16(lds_off(sA), A + (size_t)mBase * K + k0,
                               /*width=*/64, /*rows=*/128, /*stride=*/K);
        }
        // stage W tile 64x64, transposed into sBt [n][k]
#pragma unroll
        for (int i = 0; i < 2; ++i) {
            int idx = tid * 2 + i;
            int row = idx >> 3, seg = idx & 7;
            Pack8 p;
            p.u = *(const u32x4*)(W + (size_t)(k0 + row) * N + nBase + seg * 8);
#pragma unroll
            for (int t = 0; t < 8; ++t) sBt[(seg * 8 + t) * 64 + row] = p.e[t];
            if (k0 + 64 < K)
                __builtin_prefetch(W + (size_t)(k0 + 64 + row) * N + nBase + seg * 8, 0, 0);
        }
        if (wave == 0) __builtin_amdgcn_s_wait_tensorcnt(0);
        __syncthreads();
#pragma unroll
        for (int kk = 0; kk < 2; ++kk) {
            v16bf a0 = load_frag_a(sA + (waveM * 32) * 64 + kk * 32, 64);
            v16bf a1 = load_frag_a(sA + (waveM * 32 + 16) * 64 + kk * 32, 64);
            v16bf b0 = load_frag_b(sBt + (waveN * 32) * 64 + kk * 32, 64);
            v16bf b1 = load_frag_b(sBt + (waveN * 32 + 16) * 64 + kk * 32, 64);
            c[0][0] = wmma_bf16(a0, b0, c[0][0]);
            c[0][1] = wmma_bf16(a0, b1, c[0][1]);
            c[1][0] = wmma_bf16(a1, b0, c[1][0]);
            c[1][1] = wmma_bf16(a1, b1, c[1][1]);
        }
        __syncthreads();
    }

    const int hl = lane >> 4, nl = lane & 15;
#pragma unroll
    for (int i = 0; i < 2; ++i) {
#pragma unroll
        for (int j = 0; j < 2; ++j) {
            int col = nBase + waveN * 32 + j * 16 + nl;
            float bv = bias ? bias[col] : 0.f;
#pragma unroll
            for (int r = 0; r < 8; ++r) {
                int rowg = mBase + waveM * 32 + i * 16 + hl * 8 + r;
                float val = c[i][j][r] + bv;
                size_t off = (size_t)rowg * N + col;
                if (MODE == 0) outF[off] = val;
                else if (MODE == 1) outF[off] += val;
                else if (MODE == 2) outB[off] = (bf16)val;
                else outB[off] = (bf16)(val > 0.f ? val : 0.f);
            }
        }
    }
}

// ---------------------------------------------------------------------------
// Flash attention: per (b,h), 4 waves/block, 16 query rows per wave,
// stream keys in chunks of 32 through LDS. qkv[tok,1536] bf16 layout:
// q at col h*64, k at 512+h*64, v at 1024+h*64.
// ---------------------------------------------------------------------------
__global__ __launch_bounds__(128)
void attention_kernel(const bf16* __restrict__ qkv, const float* __restrict__ btab,
                      const unsigned char* __restrict__ pad,
                      bf16* __restrict__ obuf) {
    const int bh = blockIdx.x;
    const int b = bh >> 3, h = bh & 7;
    const int wave = threadIdx.x >> 5, lane = threadIdx.x & 31;
    const int hl = lane >> 4, nl = lane & 15;
    const int qbase = blockIdx.y * 64 + wave * 16;

    __shared__ __align__(16) bf16 sK[32 * 64];       // [key][dh]
    __shared__ __align__(16) bf16 sVt[64 * 32];      // [dh][key]
    __shared__ __align__(16) bf16 sP[4 * 16 * 32];   // per-wave P tile
    bf16* myP = sP + wave * (16 * 32);

    // preload Q fragments (16 rows x 64 dh -> two K=32 A-frags), direct global
    const bf16* qptr = qkv + ((size_t)(b * LL + qbase)) * (3 * DD) + h * DH;
    v16bf qf[2];
#pragma unroll
    for (int kk = 0; kk < 2; ++kk) qf[kk] = load_frag_a(qptr + kk * 32, 3 * DD);

    v8f o[4] = {};
    float mrow[8], lrow[8];
#pragma unroll
    for (int r = 0; r < 8; ++r) { mrow[r] = -1e30f; lrow[r] = 0.f; }

    const float scale = 0.125f;   // 1/sqrt(64)

    for (int c = 0; c < LL / 32; ++c) {
        // stage K chunk via async global->LDS; V chunk transposed via registers
#pragma unroll
        for (int i = 0; i < 2; ++i) {
            int idx = threadIdx.x * 2 + i;
            int row = idx >> 3, seg = idx & 7;
            const bf16* kg = qkv + ((size_t)(b * LL + c * 32 + row)) * (3 * DD) +
                             DD + h * DH + seg * 8;
            async_copy_b128(lds_off(sK + row * 64 + seg * 8), kg);
            const bf16* vg = qkv + ((size_t)(b * LL + c * 32 + row)) * (3 * DD) +
                             2 * DD + h * DH + seg * 8;
            Pack8 p;
            p.u = *(const u32x4*)vg;
#pragma unroll
            for (int tt = 0; tt < 8; ++tt) sVt[(seg * 8 + tt) * 32 + row] = p.e[tt];
        }
        async_wait0();
        __syncthreads();

        // S = Q * K^T (two key tiles of 16)
        v8f s0 = {}, s1 = {};
#pragma unroll
        for (int kk = 0; kk < 2; ++kk) {
            v16bf bk0 = load_frag_b(sK + kk * 32, 64);
            v16bf bk1 = load_frag_b(sK + 16 * 64 + kk * 32, 64);
            s0 = wmma_bf16(qf[kk], bk0, s0);
            s1 = wmma_bf16(qf[kk], bk1, s1);
        }

        // bias + padding mask + online softmax (DPP butterfly reductions)
        const int k0g = c * 32 + nl, k1g = k0g + 16;
        const float pm0 = pad[b * LL + k0g] ? -1e9f : 0.f;
        const float pm1 = pad[b * LL + k1g] ? -1e9f : 0.f;
#pragma unroll
        for (int r = 0; r < 8; ++r) {
            int qg = qbase + hl * 8 + r;
            float v0 = s0[r] * scale + btab[qg * LL + k0g] + pm0;
            float v1 = s1[r] * scale + btab[qg * LL + k1g] + pm1;
            float mx = red16_max(fmaxf(v0, v1));
            float mnew = fmaxf(mrow[r], mx);
            float alpha = __expf(mrow[r] - mnew);
            float p0 = __expf(v0 - mnew), p1 = __expf(v1 - mnew);
            float rs = red16_sum(p0 + p1);
            lrow[r] = lrow[r] * alpha + rs;
            mrow[r] = mnew;
#pragma unroll
            for (int t2 = 0; t2 < 4; ++t2) o[t2][r] *= alpha;
            int m = hl * 8 + r;
            myP[m * 32 + nl]      = (bf16)p0;
            myP[m * 32 + 16 + nl] = (bf16)p1;
        }
        // same-wave DS ordering: P stores must land before fragment reload
        asm volatile("s_wait_dscnt 0" ::: "memory");
        v16bf pf = load_frag_a(myP, 32);
#pragma unroll
        for (int t2 = 0; t2 < 4; ++t2) {
            v16bf bv = load_frag_b(sVt + (t2 * 16) * 32, 32);
            o[t2] = wmma_bf16(pf, bv, o[t2]);
        }
        __syncthreads();
    }

    // normalize and store O (bf16) into obuf[tok, 512]
#pragma unroll
    for (int t2 = 0; t2 < 4; ++t2) {
#pragma unroll
        for (int r = 0; r < 8; ++r) {
            int qg = qbase + hl * 8 + r;
            int col = h * DH + t2 * 16 + nl;
            obuf[(size_t)(b * LL + qg) * DD + col] = (bf16)(o[t2][r] / lrow[r]);
        }
    }
}

// ---------------------------------------------------------------------------
// masked mean pool over L
// ---------------------------------------------------------------------------
__global__ __launch_bounds__(256)
void pool_kernel(const float* __restrict__ x, const unsigned char* __restrict__ pad,
                 float* __restrict__ pooled) {
    const int b = blockIdx.x;
    for (int d = threadIdx.x; d < DD; d += 256) {
        float s = 0.f, cnt = 0.f;
        for (int l = 0; l < LL; ++l) {
            float msk = pad[b * LL + l] ? 0.f : 1.f;
            s += x[((size_t)(b * LL + l)) * DD + d] * msk;
            cnt += msk;
        }
        pooled[b * DD + d] = s / fmaxf(cnt, 1e-9f);
    }
}

// ---------------------------------------------------------------------------
// classifier: gelu(pooled @ Wc1 + bc1) @ Wc2 + bc2 ; one block per batch
// ---------------------------------------------------------------------------
__global__ __launch_bounds__(256)
void classifier_kernel(const float* __restrict__ pooled, const float* __restrict__ Wc1,
                       const float* __restrict__ bc1, const float* __restrict__ Wc2,
                       const float* __restrict__ bc2, float* __restrict__ out) {
    const int b = blockIdx.x;
    const int j = threadIdx.x;   // 256 hidden units
    __shared__ float red[256];
    float acc = bc1[j];
    for (int d = 0; d < DD; ++d) acc += pooled[b * DD + d] * Wc1[d * 256 + j];
    float g = 0.5f * acc * (1.f + erff(acc * 0.70710678118f));   // exact GELU
    red[j] = g * Wc2[j];
    __syncthreads();
    for (int o = 128; o > 0; o >>= 1) {
        if (j < o) red[j] += red[j + o];
        __syncthreads();
    }
    if (j == 0) out[b] = red[0] + bc2[0];
}

// ---------------------------------------------------------------------------
// host launcher
// ---------------------------------------------------------------------------
extern "C" void kernel_launch(void* const* d_in, const int* in_sizes, int n_in,
                              void* d_out, int out_size, void* d_ws, size_t ws_size,
                              hipStream_t stream) {
    (void)in_sizes; (void)n_in; (void)out_size; (void)ws_size;

    const int*           event_ids = (const int*)d_in[0];
    const float*         tdel      = (const float*)d_in[1];
    const float*         cont      = (const float*)d_in[2];
    const int*           pkg       = (const int*)d_in[3];
    const unsigned char* pad       = (const unsigned char*)d_in[4];
    const float* emb  = (const float*)d_in[5];
    const float* W1   = (const float*)d_in[6];
    const float* b1   = (const float*)d_in[7];
    const float* g1   = (const float*)d_in[8];
    const float* be1  = (const float*)d_in[9];
    const float* W2   = (const float*)d_in[10];
    const float* b2   = (const float*)d_in[11];
    const float* rel  = (const float*)d_in[12];
    const float* Wqkv = (const float*)d_in[13];
    const float* bqkv = (const float*)d_in[14];
    const float* Wo   = (const float*)d_in[15];
    const float* bo   = (const float*)d_in[16];
    const float* ln1g = (const float*)d_in[17];
    const float* ln1b = (const float*)d_in[18];
    const float* ln2g = (const float*)d_in[19];
    const float* ln2b = (const float*)d_in[20];
    const float* Wf1  = (const float*)d_in[21];
    const float* bf1  = (const float*)d_in[22];
    const float* Wf2  = (const float*)d_in[23];
    const float* bf2  = (const float*)d_in[24];
    const float* Wc1  = (const float*)d_in[25];
    const float* bc1  = (const float*)d_in[26];
    const float* Wc2  = (const float*)d_in[27];
    const float* bc2  = (const float*)d_in[28];
    float* logits = (float*)d_out;

    // ---- workspace carve (256B aligned slabs) ----
    size_t off = 0;
    auto carve = [&](size_t bytes) -> void* {
        void* p = (char*)d_ws + off;
        off += (bytes + 255) & ~(size_t)255;
        return p;
    };
    float* x      = (float*)carve((size_t)NTOK * DD * 4);
    bf16*  hn     = (bf16*) carve((size_t)NTOK * DD * 2);
    bf16*  qkvb   = (bf16*) carve((size_t)NTOK * 3 * DD * 2);
    bf16*  obuf   = (bf16*) carve((size_t)NTOK * DD * 2);
    bf16*  ffn    = (bf16*) carve((size_t)NTOK * FF * 2);
    float* btab   = (float*)carve((size_t)LL * LL * 4);
    float* pooled = (float*)carve((size_t)BB * DD * 4);
    bf16*  W2b    = (bf16*) carve((size_t)DD * DD * 2);
    bf16*  Wqkvb  = (bf16*) carve((size_t)NLAYER * DD * 3 * DD * 2);
    bf16*  Wob    = (bf16*) carve((size_t)NLAYER * DD * DD * 2);
    bf16*  Wf1b   = (bf16*) carve((size_t)NLAYER * DD * FF * 2);
    bf16*  Wf2b   = (bf16*) carve((size_t)NLAYER * FF * DD * 2);

    auto cvt = [&](const float* s, bf16* d, int n) {
        f32_to_bf16_kernel<<<(n + 255) / 256, 256, 0, stream>>>(s, d, n);
    };
    cvt(W2, W2b, DD * DD);
    cvt(Wqkv, Wqkvb, NLAYER * DD * 3 * DD);
    cvt(Wo, Wob, NLAYER * DD * DD);
    cvt(Wf1, Wf1b, NLAYER * DD * FF);
    cvt(Wf2, Wf2b, NLAYER * FF * DD);

    bias_tab_kernel<<<LL, 256, 0, stream>>>(rel, btab);

    // embedding front-end -> hn (bf16), then x = hn @ W2 + b2
    embed_kernel<<<NTOK, 256, 0, stream>>>(event_ids, tdel, cont, pkg, emb,
                                           W1, b1, g1, be1, hn);
    {
        dim3 g(NTOK / 128, DD / 64);
        gemm_bf16_kernel<0><<<g, 256, 0, stream>>>(hn, W2b, b2, x, (bf16*)nullptr,
                                                   NTOK, DD, DD);
    }

    for (int i = 0; i < NLAYER; ++i) {
        // LN1 -> hn, QKV projection
        ln_to_bf16_kernel<<<NTOK, 256, 0, stream>>>(x, ln1g + i * DD, ln1b + i * DD, hn);
        {
            dim3 g(NTOK / 128, (3 * DD) / 64);
            gemm_bf16_kernel<2><<<g, 256, 0, stream>>>(
                hn, Wqkvb + (size_t)i * DD * 3 * DD, bqkv + i * 3 * DD,
                (float*)nullptr, qkvb, NTOK, 3 * DD, DD);
        }
        // attention
        {
            dim3 g(BB * HH, LL / 64);
            attention_kernel<<<g, 128, 0, stream>>>(qkvb, btab, pad, obuf);
        }
        // x += O @ Wo + bo
        {
            dim3 g(NTOK / 128, DD / 64);
            gemm_bf16_kernel<1><<<g, 256, 0, stream>>>(
                obuf, Wob + (size_t)i * DD * DD, bo + i * DD,
                x, (bf16*)nullptr, NTOK, DD, DD);
        }
        // LN2 -> hn, FFN
        ln_to_bf16_kernel<<<NTOK, 256, 0, stream>>>(x, ln2g + i * DD, ln2b + i * DD, hn);
        {
            dim3 g(NTOK / 128, FF / 64);
            gemm_bf16_kernel<3><<<g, 256, 0, stream>>>(
                hn, Wf1b + (size_t)i * DD * FF, bf1 + i * FF,
                (float*)nullptr, ffn, NTOK, FF, DD);
        }
        {
            dim3 g(NTOK / 128, DD / 64);
            gemm_bf16_kernel<1><<<g, 256, 0, stream>>>(
                ffn, Wf2b + (size_t)i * FF * DD, bf2 + i * DD,
                x, (bf16*)nullptr, NTOK, DD, FF);
        }
    }

    pool_kernel<<<BB, 256, 0, stream>>>(x, pad, pooled);
    classifier_kernel<<<BB, 256, 0, stream>>>(pooled, Wc1, bc1, Wc2, bc2, logits);
}
